// ExpertChoiceFFN_17643725652421
// MI455X (gfx1250) — compile-verified, matrix operand order checked by reading, not run on hardware
//
#include <hip/hip_runtime.h>

// Problem constants (from reference): B=4, S=2048, HID=1024, DFF=4096, E=8, K=2048
constexpr int NTOK = 8192;     // B*S tokens
constexpr int HIDC = 1024;
constexpr int DFFC = 4096;
constexpr int NE   = 8;
constexpr int TOPK = 2048;
constexpr int LDST = 72;       // padded LDS stride (bf16 elems): 144B rows, 16B aligned

typedef __attribute__((ext_vector_type(16))) __bf16 v16bf;
typedef __attribute__((ext_vector_type(8)))  float  v8f;
typedef __attribute__((ext_vector_type(4)))  int    v4i;

#define DEVFN __device__ __forceinline__

#if __has_builtin(__builtin_amdgcn_global_load_async_to_lds_b128)
#define USE_ASYNC 1
#else
#define USE_ASYNC 0
#endif

DEVFN unsigned short f2bf_bits(float f) {              // round-to-nearest-even f32->bf16
  unsigned u = __float_as_uint(f);
  unsigned r = 0x7fffu + ((u >> 16) & 1u);
  return (unsigned short)((u + r) >> 16);
}
DEVFN unsigned pack2bf(float a, float b) {
  return (unsigned)f2bf_bits(a) | ((unsigned)f2bf_bits(b) << 16);
}
DEVFN unsigned keyOf(float f) {                        // monotone float->uint map
  unsigned u = __float_as_uint(f);
  return (u & 0x80000000u) ? ~u : (u | 0x80000000u);
}

#if USE_ASYNC
DEVFN void cp_async_b128(const unsigned short* g, unsigned short* l) {
  __builtin_amdgcn_global_load_async_to_lds_b128(
      (__attribute__((address_space(1))) v4i*)g,
      (__attribute__((address_space(3))) v4i*)l, 0, 0);
}
DEVFN void wait_async0() {
#if __has_builtin(__builtin_amdgcn_s_wait_asynccnt)
  __builtin_amdgcn_s_wait_asynccnt(0);
#else
  asm volatile("s_wait_asynccnt 0x0" ::: "memory");
#endif
}
#endif

// ---------------------------------------------------------------- zero output
__global__ void moe_zero_f32(float* p, int n) {
  int i = (blockIdx.x * blockDim.x + threadIdx.x) * 4;
  if (i < n) { float4 z = {0.f, 0.f, 0.f, 0.f}; *(float4*)(p + i) = z; }
}

// ---------------------------------------------------------------- router + softmax
// one wave (32 lanes) per token; writes scoresT[e*NTOK + tok]
__global__ void moe_router(const float* __restrict__ x, const float* __restrict__ rw,
                           const float* __restrict__ rb, float* __restrict__ scoresT) {
  int wave = threadIdx.x >> 5, lane = threadIdx.x & 31;
  int tok = blockIdx.x * 8 + wave;
  const float* xr = x + (size_t)tok * HIDC;
  float acc[NE];
#pragma unroll
  for (int e = 0; e < NE; ++e) acc[e] = 0.f;
  for (int h = lane; h < HIDC; h += 32) {
    float xv = xr[h];
    const float4* w4 = (const float4*)(rw + h * NE);
    float4 w0 = w4[0], w1 = w4[1];
    acc[0] += xv * w0.x; acc[1] += xv * w0.y; acc[2] += xv * w0.z; acc[3] += xv * w0.w;
    acc[4] += xv * w1.x; acc[5] += xv * w1.y; acc[6] += xv * w1.z; acc[7] += xv * w1.w;
  }
#pragma unroll
  for (int off = 16; off > 0; off >>= 1)
#pragma unroll
    for (int e = 0; e < NE; ++e) acc[e] += __shfl_xor(acc[e], off, 32);
  if (lane == 0) {
    float lg[NE], m = -3.4e38f, s = 0.f;
#pragma unroll
    for (int e = 0; e < NE; ++e) { lg[e] = acc[e] + rb[e]; m = fmaxf(m, lg[e]); }
#pragma unroll
    for (int e = 0; e < NE; ++e) { lg[e] = __expf(lg[e] - m); s += lg[e]; }
    float inv = 1.f / s;
#pragma unroll
    for (int e = 0; e < NE; ++e) scoresT[(size_t)e * NTOK + tok] = lg[e] * inv;
  }
}

// ---------------------------------------------------------------- per-expert top-K
// one workgroup per expert; 4-pass radix select on 32-bit keys, then selection
__global__ void moe_topk(const float* __restrict__ scoresT, int* __restrict__ idx,
                         float* __restrict__ gate) {
  __shared__ unsigned hist[256];
  __shared__ unsigned s_prefix;
  __shared__ int s_remaining, cntA, cntB;
  const int e = blockIdx.x, tid = threadIdx.x, bd = blockDim.x;
  const float* sc = scoresT + (size_t)e * NTOK;

  unsigned prefix = 0; int remaining = TOPK;
  for (int pass = 0; pass < 4; ++pass) {
    int shift = 24 - pass * 8;
    for (int i = tid; i < 256; i += bd) hist[i] = 0u;
    __syncthreads();
    for (int t = tid; t < NTOK; t += bd) {
      unsigned k = keyOf(sc[t]);
      if (pass == 0 || (k >> (shift + 8)) == prefix)
        atomicAdd(&hist[(k >> shift) & 255u], 1u);
    }
    __syncthreads();
    if (tid == 0) {
      int cum = 0, b = 255;
      for (; b > 0; --b) {
        int c = (int)hist[b];
        if (cum + c >= remaining) break;
        cum += c;
      }
      s_prefix = (prefix << 8) | (unsigned)b;
      s_remaining = remaining - cum;
    }
    __syncthreads();
    prefix = s_prefix; remaining = s_remaining;
    __syncthreads();
  }
  const unsigned thr = prefix;          // full 32-bit threshold key
  const int numEq = remaining;          // how many == thr to take
  const int cntG  = TOPK - numEq;       // count of strictly-greater keys

  if (tid == 0) { cntA = 0; cntB = 0; }
  __syncthreads();
  for (int t = tid; t < NTOK; t += bd) {
    float s = sc[t];
    unsigned k = keyOf(s);
    if (k > thr) {
      int p = atomicAdd(&cntA, 1);
      idx[e * TOPK + p] = t; gate[e * TOPK + p] = s;
    } else if (k == thr) {
      int p = atomicAdd(&cntB, 1);
      if (p < numEq) {
        int slot = cntG + p;
        idx[e * TOPK + slot] = t; gate[e * TOPK + slot] = s;
      }
    }
  }
}

// ---------------------------------------------------------------- gather + cast x rows
__global__ void moe_gather_cast(const float* __restrict__ x, const int* __restrict__ idxE,
                                unsigned short* __restrict__ toksb) {
  int k = blockIdx.x;
  int t = idxE[k];
  const float* src = x + (size_t)t * HIDC;
  unsigned short* dst = toksb + (size_t)k * HIDC;
  int h = threadIdx.x * 4;                        // 256 threads * 4 = 1024
  float4 v = *(const float4*)(src + h);
  uint2 o = { pack2bf(v.x, v.y), pack2bf(v.z, v.w) };
  *(uint2*)(dst + h) = o;
}

// ---------------------------------------------------------------- f32 -> bf16 transpose
// src [R][C] f32  ->  dst [C][R] bf16  (32x32 LDS tiles; 256 threads = 32x8)
__global__ void moe_cast_transpose(const float* __restrict__ src,
                                   unsigned short* __restrict__ dst, int R, int C) {
  __shared__ unsigned short tile[32][33];
  const int c0 = blockIdx.x * 32, r0 = blockIdx.y * 32;
  const int tx = threadIdx.x & 31, ty = threadIdx.x >> 5;
#pragma unroll
  for (int i = ty; i < 32; i += 8)
    tile[i][tx] = f2bf_bits(src[(size_t)(r0 + i) * C + c0 + tx]);
  __syncthreads();
#pragma unroll
  for (int i = ty; i < 32; i += 8)
    dst[(size_t)(c0 + i) * R + r0 + tx] = tile[tx][i];
}

// ---------------------------------------------------------------- tiled bf16 WMMA GEMM
// C[M,N] = A[M,Kd] * B[Kd,N] with B provided TRANSPOSED (BT[N][Kd], row-major).
// Block 128x128, k-tile 64, 4 waves in 2x2 grid, 64x64 per wave (4x4 wmma accums).
// Double-buffered LDS; async global->LDS staging when available.
// EPI==0: out = relu(C + bias) -> bf16 store (h1)
// EPI==1: atomicAdd y[idxE[row]*HID + col] += gateE[row] * (C + bias)
union Frag { v16bf v; uint4 q[2]; };

template <int EPI>
__global__ __launch_bounds__(128)
void moe_gemm_bf16(const unsigned short* __restrict__ A, const unsigned short* __restrict__ BT,
                   int M, int N, int Kd, const float* __restrict__ bias,
                   const int* __restrict__ idxE, const float* __restrict__ gateE,
                   unsigned short* __restrict__ outBf, float* __restrict__ outY) {
  __shared__ unsigned short ldsA[2][128 * LDST];
  __shared__ unsigned short ldsB[2][128 * LDST];

  const int blockN = blockIdx.x * 128;
  const int blockM = blockIdx.y * 128;
  const int tid  = threadIdx.x;
  const int wave = tid >> 5, lane = tid & 31;
  const int wm = wave >> 1, wn = wave & 1;         // 2 x 2 wave grid, 64x64 each
  const int lr = lane & 15;
  const int khalf = (lane >> 4) << 3;              // +0 / +8 K offset per lane half

  const int ktiles = Kd >> 6;

  // stage one 128x64 bf16 tile of A and of BT into LDS buffer `buf`
  auto stage = [&](int kt, int buf) {
    const unsigned short* gA = A  + (size_t)blockM * Kd + kt * 64;
    const unsigned short* gB = BT + (size_t)blockN * Kd + kt * 64;
#if USE_ASYNC
#pragma unroll
    for (int i = 0; i < 8; ++i) {
      int lin = tid + i * 128;                     // 0..1023
      int row = lin >> 3, k8 = (lin & 7) << 3;
      cp_async_b128(gA + (size_t)row * Kd + k8, &ldsA[buf][row * LDST + k8]);
      cp_async_b128(gB + (size_t)row * Kd + k8, &ldsB[buf][row * LDST + k8]);
    }
#else
    uint4 ra[8], rb[8];
#pragma unroll
    for (int i = 0; i < 8; ++i) {
      int lin = tid + i * 128;
      int row = lin >> 3, k8 = (lin & 7) << 3;
      ra[i] = *(const uint4*)(gA + (size_t)row * Kd + k8);
      rb[i] = *(const uint4*)(gB + (size_t)row * Kd + k8);
    }
#pragma unroll
    for (int i = 0; i < 8; ++i) {
      int lin = tid + i * 128;
      int row = lin >> 3, k8 = (lin & 7) << 3;
      *(uint4*)(&ldsA[buf][row * LDST + k8]) = ra[i];
      *(uint4*)(&ldsB[buf][row * LDST + k8]) = rb[i];
    }
#endif
  };

  v8f cacc[4][4];
  v8f vz = {0.f, 0.f, 0.f, 0.f, 0.f, 0.f, 0.f, 0.f};
#pragma unroll
  for (int mi = 0; mi < 4; ++mi)
#pragma unroll
    for (int ni = 0; ni < 4; ++ni) cacc[mi][ni] = vz;

  stage(0, 0);
  for (int kt = 0; kt < ktiles; ++kt) {
    const int buf = kt & 1;
#if USE_ASYNC
    wait_async0();                                 // my async copies for `buf` landed
#endif
    __syncthreads();                               // everyone's landed; prev reads done
    if (kt + 1 < ktiles) stage(kt + 1, buf ^ 1);   // prefetch overlaps compute below

#pragma unroll
    for (int kk = 0; kk < 64; kk += 32) {
      const int kb = kk + khalf;
      Frag af[4], bq[4];
#pragma unroll
      for (int mi = 0; mi < 4; ++mi) {
        const unsigned short* p = &ldsA[buf][(wm * 64 + mi * 16 + lr) * LDST + kb];
        af[mi].q[0] = *(const uint4*)p;            // K kb..kb+7
        af[mi].q[1] = *(const uint4*)(p + 16);     // K kb+16..kb+23
      }
#pragma unroll
      for (int ni = 0; ni < 4; ++ni) {
        const unsigned short* p = &ldsB[buf][(wn * 64 + ni * 16 + lr) * LDST + kb];
        bq[ni].q[0] = *(const uint4*)p;
        bq[ni].q[1] = *(const uint4*)(p + 16);
      }
#pragma unroll
      for (int mi = 0; mi < 4; ++mi)
#pragma unroll
        for (int ni = 0; ni < 4; ++ni)
          cacc[mi][ni] = __builtin_amdgcn_wmma_f32_16x16x32_bf16(
              false, af[mi].v, false, bq[ni].v, (short)0, cacc[mi][ni], false, false);
    }
  }

  // ---- epilogue (C layout: VGPR r -> M = r (lanes 0-15) or r+8 (lanes 16-31); N = lane&15)
#pragma unroll
  for (int mi = 0; mi < 4; ++mi)
#pragma unroll
    for (int ni = 0; ni < 4; ++ni) {
      const int col = blockN + wn * 64 + ni * 16 + lr;
      const float bv = bias[col];
#pragma unroll
      for (int r = 0; r < 8; ++r) {
        const int row = blockM + wm * 64 + mi * 16 + ((lane < 16) ? r : r + 8);
        float v = cacc[mi][ni][r] + bv;
        if constexpr (EPI == 0) {
          v = fmaxf(v, 0.f);
          outBf[(size_t)row * N + col] = f2bf_bits(v);
        } else {
          const int token = idxE[row];
          const float g = gateE[row];
          unsafeAtomicAdd(outY + (size_t)token * HIDC + col, v * g);
        }
      }
    }
}

// ---------------------------------------------------------------- launch
extern "C" void kernel_launch(void* const* d_in, const int* in_sizes, int n_in,
                              void* d_out, int out_size, void* d_ws, size_t ws_size,
                              hipStream_t stream) {
  (void)in_sizes; (void)n_in; (void)out_size; (void)ws_size;
  const float* x  = (const float*)d_in[0];
  const float* rw = (const float*)d_in[1];
  const float* rb = (const float*)d_in[2];
  const float* w1 = (const float*)d_in[3];
  const float* b1 = (const float*)d_in[4];
  const float* w2 = (const float*)d_in[5];
  const float* b2 = (const float*)d_in[6];
  float* y = (float*)d_out;

  char* ws = (char*)d_ws;
  size_t off = 0;
  auto take = [&](size_t bytes) -> void* {
    void* p = ws + off;
    off += (bytes + 255) & ~(size_t)255;
    return p;
  };
  float*          scoresT = (float*)take((size_t)NE * NTOK * 4);            // 256 KB
  int*            idx     = (int*)take((size_t)NE * TOPK * 4);              //  64 KB
  float*          gate    = (float*)take((size_t)NE * TOPK * 4);            //  64 KB
  unsigned short* toksb   = (unsigned short*)take((size_t)TOPK * HIDC * 2); //   4 MB
  unsigned short* w1bT    = (unsigned short*)take((size_t)HIDC * DFFC * 2); //   8 MB [DFF][HID]
  unsigned short* w2bT    = (unsigned short*)take((size_t)DFFC * HIDC * 2); //   8 MB [HID][DFF]
  unsigned short* h1b     = (unsigned short*)take((size_t)TOPK * DFFC * 2); //  16 MB

  moe_zero_f32<<<(NTOK * HIDC) / 1024, 256, 0, stream>>>(y, NTOK * HIDC);
  moe_router<<<NTOK / 8, 256, 0, stream>>>(x, rw, rb, scoresT);
  moe_topk<<<NE, 256, 0, stream>>>(scoresT, idx, gate);

  for (int e = 0; e < NE; ++e) {
    moe_gather_cast<<<TOPK, 256, 0, stream>>>(x, idx + e * TOPK, toksb);
    // w1[e] [HID][DFF] -> w1bT [DFF][HID];  w2[e] [DFF][HID] -> w2bT [HID][DFF]
    moe_cast_transpose<<<dim3(DFFC / 32, HIDC / 32), 256, 0, stream>>>(
        w1 + (size_t)e * HIDC * DFFC, w1bT, HIDC, DFFC);
    moe_cast_transpose<<<dim3(HIDC / 32, DFFC / 32), 256, 0, stream>>>(
        w2 + (size_t)e * DFFC * HIDC, w2bT, DFFC, HIDC);
    // h1 = relu(toks @ w1 + b1)   [2048 x 4096]
    moe_gemm_bf16<0><<<dim3(DFFC / 128, TOPK / 128), 128, 0, stream>>>(
        toksb, w1bT, TOPK, DFFC, HIDC, b1 + (size_t)e * DFFC,
        nullptr, nullptr, h1b, nullptr);
    // y[idx] += G * (h1 @ w2 + b2)   [2048 x 1024] scatter
    moe_gemm_bf16<1><<<dim3(HIDC / 128, TOPK / 128), 128, 0, stream>>>(
        h1b, w2bT, TOPK, HIDC, DFFC, b2 + (size_t)e * HIDC,
        idx + e * TOPK, gate + e * TOPK, nullptr, y);
  }
}
